// AttCM_67396626809426
// MI455X (gfx1250) — matrix operand order repaired and checked by reading, not compiled, not run on hardware
//
#include <hip/hip_runtime.h>
#include <hip/hip_bf16.h>
#include <math.h>

// ---------------------------------------------------------------------------
// CDNA5 (gfx1250) WMMA types / helpers
// ---------------------------------------------------------------------------
typedef __attribute__((ext_vector_type(16))) _Float16 v16h;
typedef __attribute__((ext_vector_type(8)))  _Float16 v8h;   // one b128 of f16
typedef __attribute__((ext_vector_type(8)))  float    v8f;
typedef __attribute__((ext_vector_type(4)))  float    v4f;   // one b128 of f32
typedef __attribute__((ext_vector_type(4)))  unsigned v4u;
typedef __attribute__((ext_vector_type(8)))  int      v8i;
typedef __attribute__((ext_vector_type(4)))  int      v4i;

__device__ __forceinline__ v8f wmma_f16(v16h a, v16h b, v8f c) {
    return __builtin_amdgcn_wmma_f32_16x16x32_f16(false, a, false, b,
                                                  (short)0, c, false, false);
}

// A-fragment (16x32 f16, row-per-lane): lane half h -> K = h*16 + 0..15
__device__ __forceinline__ v16h load_a(const _Float16* base, int half) {
    v8h lo = *(const v8h*)(base + half * 16);
    v8h hi = *(const v8h*)(base + half * 16 + 8);
    return __builtin_shufflevector(lo, hi, 0, 1, 2, 3, 4, 5, 6, 7,
                                   8, 9, 10, 11, 12, 13, 14, 15);
}
// B-fragment (32x16 f16): lane half h -> K = h*16 + 0..15 (K-contiguous layout)
__device__ __forceinline__ v16h load_b(const _Float16* base, int half) {
    v8h lo = ((const v8h*)(base + half * 16))[0];
    v8h hi = ((const v8h*)(base + half * 16))[1];
    return __builtin_shufflevector(lo, hi, 0, 1, 2, 3, 4, 5, 6, 7,
                                   8, 9, 10, 11, 12, 13, 14, 15);
}

// ---------------------------------------------------------------------------
// Tensor Data Mover: build a D# (ISA 8.3/8.4) and issue TENSOR_LOAD_TO_LDS.
// Up-to-3D tile: X contiguous (dim0), Y rows (stride0), Z planes (stride1).
// All dims/strides in data_size units. Issue from ONE wave, then
// s_wait_tensorcnt 0 + workgroup barrier before LDS is read.
// ---------------------------------------------------------------------------
__device__ __forceinline__ void tdm_load_to_lds(
        unsigned lds_off, const void* gptr, unsigned data_size_log2,
        unsigned td0, unsigned td1, unsigned td2,
        unsigned tile0, unsigned tile1, unsigned tile2,
        unsigned long long stride0, unsigned long long stride1) {
    const unsigned long long ga = (unsigned long long)(uintptr_t)gptr;
    v4u g0;
    g0[0] = 1u;                                   // count=1, user descriptor
    g0[1] = lds_off;                              // lds_addr [63:32]
    g0[2] = (unsigned)(ga & 0xFFFFFFFFu);         // global_addr lo
    g0[3] = (unsigned)((ga >> 32) & 0x1FFFFFFu)   // global_addr hi (57-bit)
          | (2u << 30);                           // type=2 ("image")
    v8i g1;
    g1[0] = (int)(data_size_log2 << 16);          // wg_mask=0, data_size
    g1[1] = (int)((td0 & 0xFFFFu) << 16);         // tensor_dim0[15:0]
    g1[2] = (int)((td0 >> 16) | ((td1 & 0xFFFFu) << 16));
    g1[3] = (int)((td1 >> 16) | (tile0 << 16));   // tile_dim0
    g1[4] = (int)(tile1 | (tile2 << 16));         // tile_dim1, tile_dim2
    g1[5] = (int)(stride0 & 0xFFFFFFFFull);       // tensor_dim0_stride[31:0]
    g1[6] = (int)(((stride0 >> 32) & 0xFFFFull) | ((stride1 & 0xFFFFull) << 16));
    g1[7] = (int)((stride1 >> 16) & 0xFFFFFFFFull);
    v4i g2 = { (int)td2, 0, 0, 0 };               // tensor_dim2
    v4i g3 = { 0, 0, 0, 0 };
#if __clang_major__ >= 23
    v8i g4 = { 0, 0, 0, 0, 0, 0, 0, 0 };
    __builtin_amdgcn_tensor_load_to_lds(g0, g1, g2, g3, g4, 0);
#else
    __builtin_amdgcn_tensor_load_to_lds(g0, g1, g2, g3, 0);
#endif
}

__device__ __forceinline__ unsigned lds_offset(const void* shared_ptr) {
    // Generic LDS address = SHARED_BASE aperture | 32-bit LDS offset.
    return (unsigned)(uintptr_t)shared_ptr;
}

// ---------------------------------------------------------------------------
// Data prep kernels
// ---------------------------------------------------------------------------
__global__ void cast_f32_f16_kernel(const float* __restrict__ in,
                                    _Float16* __restrict__ out, size_t n) {
    size_t i = (size_t)blockIdx.x * blockDim.x + threadIdx.x;
    if (i < n) out[i] = (_Float16)in[i];
}

// x (B,C,N) f32 -> xt (B,N,C) f16   (coalesced reads)
__global__ void transpose_cast_kernel(const float* __restrict__ x,
                                      _Float16* __restrict__ xt, int C, int N) {
    const int n = blockIdx.x * blockDim.x + threadIdx.x;
    const int c = blockIdx.y;
    const int b = blockIdx.z;
    xt[((size_t)b * N + n) * C + c] = (_Float16)x[((size_t)b * C + c) * N + n];
}

// w (Co,Ci,3,3) f32 -> o (3,3,Co,Ci) f16 so A-fragments are K-contiguous
__global__ void repack_w3x3_kernel(const float* __restrict__ w,
                                   _Float16* __restrict__ o, int Co, int Ci) {
    size_t i = (size_t)blockIdx.x * blockDim.x + threadIdx.x;
    const size_t total = (size_t)Co * Ci * 9;
    if (i >= total) return;
    const int kx = (int)(i % 3); size_t r = i / 3;
    const int ky = (int)(r % 3); r /= 3;
    const int ci = (int)(r % Ci);
    const int co = (int)(r / Ci);
    o[(((size_t)(ky * 3 + kx) * Co + co) * Ci) + ci] = (_Float16)w[i];
}

// ---------------------------------------------------------------------------
// 1x1 conv GEMM. A tile (16 x Ci weights) staged in LDS once per block by TDM;
// 8 waves x (16co x 32n) tiles share it. grid = (N/256, Co/16, B), block 256.
// ---------------------------------------------------------------------------
__global__ void __launch_bounds__(256)
gemm1x1_kernel(const _Float16* __restrict__ XT, // (B,N,Ci) f16
               const _Float16* __restrict__ Wh, // (Co,Ci)   f16
               const float* __restrict__ bias,  // (Co)
               _Float16* __restrict__ YT,       // (B,N,Co) f16 (or null)
               _Float16* __restrict__ Ycn,      // (B,Co,N) f16 (or null; V layout)
               int Ci, int Co, int N, int relu) {
    __shared__ _Float16 sW[16 * 256];
    const int lane = threadIdx.x & 31;
    const int wv   = threadIdx.x >> 5;
    const int half = lane >> 4;
    const int t    = lane & 15;
    const int nb   = (blockIdx.x * 8 + wv) * 32;
    const int cob  = blockIdx.y * 16;
    const int b    = blockIdx.z;

    if (wv == 0) { // one TDM op stages the whole 16-row weight tile
        tdm_load_to_lds(lds_offset(sW), Wh + (size_t)cob * Ci, /*2B*/1,
                        Ci, 16, 0, Ci, 16, 0, Ci, 0);
        __builtin_amdgcn_s_wait_tensorcnt(0);
    }
    __syncthreads();

    const _Float16* wrow  = sW + (size_t)t * Ci;            // LDS
    const _Float16* xrow0 = XT + ((size_t)b * N + nb + t) * Ci;
    const _Float16* xrow1 = XT + ((size_t)b * N + nb + 16 + t) * Ci;

    v8f acc[2] = {{}, {}};
#pragma unroll 2
    for (int k0 = 0; k0 < Ci; k0 += 32) {
        const v16h a  = load_a(wrow + k0, half);
        const v16h b0 = load_b(xrow0 + k0, half);
        const v16h b1 = load_b(xrow1 + k0, half);
        acc[0] = wmma_f16(a, b0, acc[0]);
        acc[1] = wmma_f16(a, b1, acc[1]);
    }
#pragma unroll
    for (int s = 0; s < 2; ++s) {
        const int n = nb + s * 16 + t;
        v8h o;
#pragma unroll
        for (int r = 0; r < 8; ++r) {
            float v = acc[s][r] + bias[cob + 8 * half + r];
            if (relu) v = fmaxf(v, 0.f);
            o[r] = (_Float16)v;
        }
        if (YT)
            *(v8h*)(YT + ((size_t)b * N + n) * Co + cob + 8 * half) = o;
        if (Ycn) {
#pragma unroll
            for (int r = 0; r < 8; ++r)
                Ycn[((size_t)b * Co + (cob + 8 * half + r)) * N + n] = o[r];
        }
    }
}

// ---------------------------------------------------------------------------
// 3x3 conv (pad=1): 9 shifted GEMMs. All 9 weight planes (16 x Ci each) staged
// by a single 3D TDM load (plane stride Co*Ci) into dynamic LDS (72 KB).
// ---------------------------------------------------------------------------
__global__ void __launch_bounds__(256)
conv3x3_kernel(const _Float16* __restrict__ XT, // (B,N,Ci) f16
               const _Float16* __restrict__ Wp, // (3,3,Co,Ci) f16
               const float* __restrict__ bias,
               _Float16* __restrict__ YT,  // (B,N,Co) f16 (or null)
               float* __restrict__ YTf,    // (B,N,Co) f32 (or null)
               int Ci, int Co, int H, int Wd, int relu) {
    extern __shared__ _Float16 sW9[]; // 9 * 16 * Ci f16
    const int lane = threadIdx.x & 31;
    const int wv   = threadIdx.x >> 5;
    const int half = lane >> 4;
    const int t    = lane & 15;
    const int N    = H * Wd;
    const int nb   = (blockIdx.x * 8 + wv) * 32;
    const int cob  = blockIdx.y * 16;
    const int b    = blockIdx.z;
    const int h    = nb / Wd;
    const int wb   = nb % Wd;
    const _Float16* Xb = XT + (size_t)b * N * Ci;

    if (wv == 0) { // 3D tile: X=Ci, Y=16 rows (stride Ci), Z=9 taps (stride Co*Ci)
        tdm_load_to_lds(lds_offset(sW9), Wp + (size_t)cob * Ci, /*2B*/1,
                        Ci, 16, 9, Ci, 16, 9,
                        (unsigned long long)Ci, (unsigned long long)Co * Ci);
        __builtin_amdgcn_s_wait_tensorcnt(0);
    }
    __syncthreads();

    v8f acc[2] = {{}, {}};
    for (int dy = -1; dy <= 1; ++dy) {
        const int hh = h + dy;
        if (hh < 0 || hh >= H) continue; // uniform across wave
        for (int dx = -1; dx <= 1; ++dx) {
            const int plane = (dy + 1) * 3 + (dx + 1);
            const _Float16* wrow = sW9 + ((size_t)plane * 16 + t) * Ci; // LDS
            const int col0 = wb + t + dx;
            const int col1 = wb + 16 + t + dx;
            const bool ok0 = (col0 >= 0) && (col0 < Wd);
            const bool ok1 = (col1 >= 0) && (col1 < Wd);
            const _Float16* xr0 = Xb + ((size_t)hh * Wd + col0) * Ci;
            const _Float16* xr1 = Xb + ((size_t)hh * Wd + col1) * Ci;
            for (int k0 = 0; k0 < Ci; k0 += 32) {
                const v16h a = load_a(wrow + k0, half);
                v16h b0 = {}, b1 = {};
                if (ok0) b0 = load_b(xr0 + k0, half);
                if (ok1) b1 = load_b(xr1 + k0, half);
                acc[0] = wmma_f16(a, b0, acc[0]);
                acc[1] = wmma_f16(a, b1, acc[1]);
            }
        }
    }
#pragma unroll
    for (int s = 0; s < 2; ++s) {
        const int n = nb + s * 16 + t;
        float v[8];
#pragma unroll
        for (int r = 0; r < 8; ++r) {
            v[r] = acc[s][r] + bias[cob + 8 * half + r];
            if (relu) v[r] = fmaxf(v[r], 0.f);
        }
        if (YT) {
            v8h o;
#pragma unroll
            for (int r = 0; r < 8; ++r) o[r] = (_Float16)v[r];
            *(v8h*)(YT + ((size_t)b * N + n) * Co + cob + 8 * half) = o;
        }
        if (YTf) {
            v4f o0, o1;
#pragma unroll
            for (int r = 0; r < 4; ++r) { o0[r] = v[r]; o1[r] = v[r + 4]; }
            float* p = YTf + ((size_t)b * N + n) * Co + cob + 8 * half;
            ((v4f*)p)[0] = o0;
            ((v4f*)p)[1] = o1;
        }
    }
}

// ---------------------------------------------------------------------------
// Attention scores, stored TRANSPOSED: ST[m,j] = sum_c K[c,j]*Q[c,m]
// Shared A tile (16 K-rows x C) staged in LDS by TDM.
// ---------------------------------------------------------------------------
__global__ void __launch_bounds__(256)
scores_kernel(const _Float16* __restrict__ KT, // (N,C)
              const _Float16* __restrict__ QT, // (N,C)
              float* __restrict__ ST,          // (N_m, N_j)
              int C, int N) {
    __shared__ _Float16 sK[16 * 256];
    const int lane = threadIdx.x & 31;
    const int wv   = threadIdx.x >> 5;
    const int half = lane >> 4;
    const int t    = lane & 15;
    const int mb   = (blockIdx.x * 8 + wv) * 32; // q-position (softmax axis)
    const int jb   = blockIdx.y * 16;            // k-position

    if (wv == 0) {
        tdm_load_to_lds(lds_offset(sK), KT + (size_t)jb * C, /*2B*/1,
                        C, 16, 0, C, 16, 0, C, 0);
        __builtin_amdgcn_s_wait_tensorcnt(0);
    }
    __syncthreads();

    const _Float16* krow = sK + (size_t)t * C;          // A[j][c] from LDS
    const _Float16* qr0  = QT + (size_t)(mb + t) * C;   // B[c][m]
    const _Float16* qr1  = QT + (size_t)(mb + 16 + t) * C;

    v8f acc[2] = {{}, {}};
#pragma unroll
    for (int k0 = 0; k0 < 256; k0 += 32) {
        const v16h a  = load_a(krow + k0, half);
        const v16h b0 = load_b(qr0 + k0, half);
        const v16h b1 = load_b(qr1 + k0, half);
        acc[0] = wmma_f16(a, b0, acc[0]);
        acc[1] = wmma_f16(a, b1, acc[1]);
    }
#pragma unroll
    for (int s = 0; s < 2; ++s) {
        const int m = mb + s * 16 + t;
        v4f o0, o1;
#pragma unroll
        for (int r = 0; r < 4; ++r) { o0[r] = acc[s][r]; o1[r] = acc[s][r + 4]; }
        float* p = ST + (size_t)m * N + jb + 8 * half;
        ((v4f*)p)[0] = o0;
        ((v4f*)p)[1] = o1;
    }
}

// ---------------------------------------------------------------------------
// Softmax row stats over ST columns (coalesced across j)
// ---------------------------------------------------------------------------
__global__ void rowstats_kernel(const float* __restrict__ ST,
                                float* __restrict__ mx, float* __restrict__ rl,
                                int N) {
    const int j = blockIdx.x * blockDim.x + threadIdx.x;
    float m = -1e30f;
    for (int n = 0; n < N; ++n) m = fmaxf(m, ST[(size_t)n * N + j]);
    float l = 0.f;
    for (int n = 0; n < N; ++n) l += __expf(ST[(size_t)n * N + j] - m);
    mx[j] = m;
    rl[j] = 1.f / l;
}

// P-fragment: B[j][m] tile of P = exp(ST[m,j]-mx[j])*rl[j]; stats from LDS.
__device__ __forceinline__ v16h p_frag(const float* __restrict__ srow,
                                       const float* smx, const float* srl,
                                       int k0, int half) {
    const int j0 = k0 + half * 16;
    v4f s[4], mv[4], rv[4];
#pragma unroll
    for (int q = 0; q < 4; ++q) {
        s[q]  = ((const v4f*)(srow + j0))[q];
        mv[q] = ((const v4f*)(smx + j0))[q];
        rv[q] = ((const v4f*)(srl + j0))[q];
    }
    v16h o;
#pragma unroll
    for (int e = 0; e < 16; ++e)
        o[e] = (_Float16)(__expf(s[e >> 2][e & 3] - mv[e >> 2][e & 3])
                          * rv[e >> 2][e & 3]);
    return o;
}

// ---------------------------------------------------------------------------
// Attention output + final combine (one batch). mx|rl (contiguous in ws)
// staged as one 2D TDM tile (N x 2) into LDS; read every K-step from LDS.
// ---------------------------------------------------------------------------
__global__ void __launch_bounds__(256)
attnout_kernel(const _Float16* __restrict__ Vf, // (C,N)
               const float* __restrict__ ST,    // (N_m, N_j)
               const float* __restrict__ mx,    // (N) ; rl = mx + N contiguous
               const float* __restrict__ convT, // (N,C) f32
               const float* __restrict__ alpha, const float* __restrict__ beta,
               float* __restrict__ out,         // (C,N) f32
               int C, int N) {
    __shared__ float sMR[2 * 4096]; // [0,N): mx, [N,2N): rl
    const int lane = threadIdx.x & 31;
    const int wv   = threadIdx.x >> 5;
    const int half = lane >> 4;
    const int t    = lane & 15;
    const int mb   = (blockIdx.x * 8 + wv) * 32;
    const int cb   = blockIdx.y * 16;

    if (wv == 0) { // 2D tile: X = N floats, Y = 2 rows (mx then rl)
        tdm_load_to_lds(lds_offset(sMR), mx, /*4B*/2,
                        N, 2, 0, N, 2, 0, (unsigned long long)N, 0);
        __builtin_amdgcn_s_wait_tensorcnt(0);
    }
    __syncthreads();
    const float* smx = sMR;
    const float* srl = sMR + N;

    const _Float16* vrow = Vf + (size_t)(cb + t) * N;   // A[c][j]
    const float* sr0 = ST + (size_t)(mb + t) * N;       // P columns m
    const float* sr1 = ST + (size_t)(mb + 16 + t) * N;

    v8f acc[2] = {{}, {}};
#pragma unroll 2
    for (int k0 = 0; k0 < N; k0 += 32) {
        if (k0 + 32 < N) { // stream-ahead (global_prefetch_b8)
            __builtin_prefetch(sr0 + k0 + 32, 0, 1);
            __builtin_prefetch(sr1 + k0 + 32, 0, 1);
        }
        const v16h a  = load_a(vrow + k0, half);
        const v16h b0 = p_frag(sr0, smx, srl, k0, half);
        const v16h b1 = p_frag(sr1, smx, srl, k0, half);
        acc[0] = wmma_f16(a, b0, acc[0]);
        acc[1] = wmma_f16(a, b1, acc[1]);
    }
    const float al = alpha[0], be = beta[0];
#pragma unroll
    for (int s = 0; s < 2; ++s) {
        const int m = mb + s * 16 + t;
        const float* cf = convT + (size_t)m * C + cb + 8 * half;
        const v4f c0 = ((const v4f*)cf)[0];
        const v4f c1 = ((const v4f*)cf)[1];
#pragma unroll
        for (int r = 0; r < 8; ++r) {
            const float cv = (r < 4) ? c0[r] : c1[r - 4];
            out[(size_t)(cb + 8 * half + r) * N + m] = al * cv + be * acc[s][r];
        }
    }
}

// ---------------------------------------------------------------------------
// Host launcher
// ---------------------------------------------------------------------------
extern "C" void kernel_launch(void* const* d_in, const int* in_sizes, int n_in,
                              void* d_out, int out_size, void* d_ws, size_t ws_size,
                              hipStream_t stream) {
    (void)in_sizes; (void)n_in; (void)out_size; (void)ws_size;

    const int B = 8, C = 256, H = 64, Wd = 64;
    const int N = H * Wd;            // 4096
    const size_t CN  = (size_t)C * N;
    const size_t BCN = (size_t)B * CN;

    const float* x   = (const float*)d_in[0];
    const float* w1  = (const float*)d_in[1];  const float* b1  = (const float*)d_in[2];
    const float* w2  = (const float*)d_in[3];  const float* b2  = (const float*)d_in[4];
    const float* w3  = (const float*)d_in[5];  const float* b3  = (const float*)d_in[6];
    const float* wb1 = (const float*)d_in[7];  const float* bb1 = (const float*)d_in[8];
    const float* wb2 = (const float*)d_in[9];  const float* bb2 = (const float*)d_in[10];
    const float* wq  = (const float*)d_in[11]; const float* bq  = (const float*)d_in[12];
    const float* wk  = (const float*)d_in[13]; const float* bk  = (const float*)d_in[14];
    const float* wv  = (const float*)d_in[15]; const float* bv  = (const float*)d_in[16];
    const float* alpha = (const float*)d_in[17];
    const float* beta  = (const float*)d_in[18];
    float* out = (float*)d_out;

    // ---- workspace carve (every chunk is a multiple of 256 B) ----
    char* p = (char*)d_ws;
    _Float16* xh  = (_Float16*)p; p += BCN * 2;                        // (B,N,256)
    _Float16* h1  = (_Float16*)p; p += (size_t)B * N * 64 * 2;         // (B,N,64)
    _Float16* h2  = (_Float16*)p; p += (size_t)B * N * 128 * 2;        // (B,N,128)
    _Float16* h3  = (_Float16*)p; p += BCN * 2;                        // (B,N,256)
    _Float16* br1 = (_Float16*)p; p += BCN * 2;                        // (B,N,256)
    _Float16* qh  = (_Float16*)p; p += BCN * 2;                        // (B,N,256)
    _Float16* kh  = (_Float16*)p; p += BCN * 2;                        // (B,N,256)
    _Float16* vh  = (_Float16*)p; p += BCN * 2;                        // (B,256,N)
    float*  convT = (float*)p;    p += BCN * 4;                        // (B,N,256) f32
    float*  ST    = (float*)p;    p += (size_t)N * N * 4;              // (N,N)
    float*  mx    = (float*)p;    p += (size_t)N * 4;                  // mx, then rl
    float*  rl    = (float*)p;    p += (size_t)N * 4;                  // contiguous!
    _Float16* w1h = (_Float16*)p; p += (size_t)64 * 256 * 2;
    _Float16* w2h = (_Float16*)p; p += (size_t)128 * 64 * 2;
    _Float16* w3h = (_Float16*)p; p += (size_t)256 * 128 * 2;
    _Float16* wqh = (_Float16*)p; p += (size_t)256 * 256 * 2;
    _Float16* wkh = (_Float16*)p; p += (size_t)256 * 256 * 2;
    _Float16* wvh = (_Float16*)p; p += (size_t)256 * 256 * 2;
    _Float16* wb1h = (_Float16*)p; p += (size_t)9 * 256 * 256 * 2;
    _Float16* wb2h = (_Float16*)p; p += (size_t)9 * 256 * 256 * 2;

    // ---- weight prep ----
    cast_f32_f16_kernel<<<(64 * 256 + 255) / 256, 256, 0, stream>>>(w1, w1h, 64 * 256);
    cast_f32_f16_kernel<<<(128 * 64 + 255) / 256, 256, 0, stream>>>(w2, w2h, 128 * 64);
    cast_f32_f16_kernel<<<(256 * 128 + 255) / 256, 256, 0, stream>>>(w3, w3h, 256 * 128);
    cast_f32_f16_kernel<<<(256 * 256 + 255) / 256, 256, 0, stream>>>(wq, wqh, 256 * 256);
    cast_f32_f16_kernel<<<(256 * 256 + 255) / 256, 256, 0, stream>>>(wk, wkh, 256 * 256);
    cast_f32_f16_kernel<<<(256 * 256 + 255) / 256, 256, 0, stream>>>(wv, wvh, 256 * 256);
    {
        size_t n = (size_t)9 * 256 * 256;
        repack_w3x3_kernel<<<(unsigned)((n + 255) / 256), 256, 0, stream>>>(wb1, wb1h, 256, 256);
        repack_w3x3_kernel<<<(unsigned)((n + 255) / 256), 256, 0, stream>>>(wb2, wb2h, 256, 256);
    }

    // ---- x -> transposed f16 ----
    transpose_cast_kernel<<<dim3(N / 256, C, B), 256, 0, stream>>>(x, xh, C, N);

    const dim3 blk(256);
    const unsigned lds3x3 = 9u * 16u * 256u * 2u; // 72 KB dynamic LDS

    // trunk of 1x1 convs (relu, transposed f16 activations)
    gemm1x1_kernel<<<dim3(N / 256, 64 / 16, B),  blk, 0, stream>>>(xh, w1h, b1, h1, nullptr, 256, 64,  N, 1);
    gemm1x1_kernel<<<dim3(N / 256, 128 / 16, B), blk, 0, stream>>>(h1, w2h, b2, h2, nullptr, 64,  128, N, 1);
    gemm1x1_kernel<<<dim3(N / 256, C / 16, B),   blk, 0, stream>>>(h2, w3h, b3, h3, nullptr, 128, C,   N, 1);

    // 3x3 conv branch: relu(conv) then conv -> f32 convT
    conv3x3_kernel<<<dim3(N / 256, C / 16, B), blk, lds3x3, stream>>>(h3,  wb1h, bb1, br1, nullptr, C, C, H, Wd, 1);
    conv3x3_kernel<<<dim3(N / 256, C / 16, B), blk, lds3x3, stream>>>(br1, wb2h, bb2, nullptr, convT, C, C, H, Wd, 0);

    // q, k (transposed) and v ((C,N) layout) 1x1 convs
    gemm1x1_kernel<<<dim3(N / 256, C / 16, B), blk, 0, stream>>>(h3, wqh, bq, qh, nullptr, C, C, N, 0);
    gemm1x1_kernel<<<dim3(N / 256, C / 16, B), blk, 0, stream>>>(h3, wkh, bk, kh, nullptr, C, C, N, 0);
    gemm1x1_kernel<<<dim3(N / 256, C / 16, B), blk, 0, stream>>>(h3, wvh, bv, nullptr, vh, C, C, N, 0);

    // attention, one batch at a time (ST is 64 MB -> fits in the 192 MB L2)
    for (int b = 0; b < B; ++b) {
        const _Float16* Kb = kh + (size_t)b * CN;
        const _Float16* Qb = qh + (size_t)b * CN;
        const _Float16* Vb = vh + (size_t)b * CN;
        scores_kernel<<<dim3(N / 256, N / 16), blk, 0, stream>>>(Kb, Qb, ST, C, N);
        rowstats_kernel<<<N / 256, 256, 0, stream>>>(ST, mx, rl, N);
        attnout_kernel<<<dim3(N / 256, C / 16), blk, 0, stream>>>(
            Vb, ST, mx, convT + (size_t)b * CN, alpha, beta,
            out + (size_t)b * CN, C, N);
    }
}
